// CrossAttentionMemory_56891136803336
// MI455X (gfx1250) — compile-verified, matrix operand order checked by reading, not run on hardware
//
#include <hip/hip_runtime.h>
#include <hip/hip_bf16.h>

typedef __attribute__((ext_vector_type(16))) __bf16 bf16x16;
typedef __attribute__((ext_vector_type(8)))  __bf16 bf16x8;
typedef __attribute__((ext_vector_type(8)))  float  floatx8;

union Frag16 { bf16x16 v; bf16x8 h[2]; };

// ---------------------------------------------------------------------------
// GEMM:  C[r, c] = scale * sum_k A[r, k] * W[c, k]      (all bf16, f32 accum)
// A: [R, K] row-major (ldA), W: [Ncol, K] row-major (ldW), C: [R, Ncol] (ldC)
// Block tile 256x128, 256 threads = 8 waves in a 4x2 grid; each wave owns a
// 64x64 tile = 4x4 WMMA 16x16x32 bf16 tiles (16 accumulators, 1 load/WMMA).
// K is processed in 128-element blocks; the next block is prefetched into
// WGP$ via global_prefetch_b8. grid.z = batch with per-operand strides.
// All dims are multiples of the tile sizes (guaranteed by launch config).
// ---------------------------------------------------------------------------
__global__ __launch_bounds__(256) void gemm_bt_wmma_kernel(
    const __bf16* __restrict__ A, const __bf16* __restrict__ W,
    __bf16* __restrict__ C, int K, int ldA, int ldW, int ldC,
    long long sA, long long sW, long long sC, float scale)
{
  const long long bz = blockIdx.z;
  A += bz * sA; W += bz * sW; C += bz * sC;

  const int tid  = threadIdx.x;
  const int wave = tid >> 5;
  const int lane = tid & 31;
  const int wm   = wave >> 1;      // 0..3
  const int wn   = wave & 1;       // 0..1
  const int rowBase = blockIdx.x * 256 + wm * 64;
  const int colBase = blockIdx.y * 128 + wn * 64;
  const int l15  = lane & 15;
  const int hi   = lane >> 4;          // 0 for lanes 0-15, 1 for 16-31
  const int kLane = hi << 3;           // +0 or +8 element K offset per lane

  // Per-lane base pointers for the 4 A-row fragments and 4 W-col fragments.
  const __bf16* aBase[4];
  const __bf16* bBase[4];
  #pragma unroll
  for (int i = 0; i < 4; ++i)
    aBase[i] = A + (long long)(rowBase + i * 16 + l15) * ldA + kLane;
  #pragma unroll
  for (int j = 0; j < 4; ++j)
    bBase[j] = W + (long long)(colBase + j * 16 + l15) * ldW + kLane;

  floatx8 acc[4][4] = {};

  for (int kb = 0; kb < K; kb += 128) {
    // Prefetch the next 128-wide K block into WGP$ (global_prefetch_b8).
    if (kb + 128 < K) {
      #pragma unroll
      for (int i = 0; i < 4; ++i) __builtin_prefetch(aBase[i] + kb + 128, 0, 0);
      #pragma unroll
      for (int j = 0; j < 4; ++j) __builtin_prefetch(bBase[j] + kb + 128, 0, 0);
    }
    #pragma unroll
    for (int kk = 0; kk < 128; kk += 32) {
      const int k0 = kb + kk;
      Frag16 a[4], b[4];
      // A fragment: lane holds row l15, 16B chunks at k0+kLane and +16 elems.
      #pragma unroll
      for (int i = 0; i < 4; ++i) {
        const __bf16* p = aBase[i] + k0;
        a[i].h[0] = *reinterpret_cast<const bf16x8*>(p);       // global_load_b128
        a[i].h[1] = *reinterpret_cast<const bf16x8*>(p + 16);  // global_load_b128
      }
      // B fragment: identical per-lane layout with column index (W is [Ncol, K]).
      #pragma unroll
      for (int j = 0; j < 4; ++j) {
        const __bf16* p = bBase[j] + k0;
        b[j].h[0] = *reinterpret_cast<const bf16x8*>(p);
        b[j].h[1] = *reinterpret_cast<const bf16x8*>(p + 16);
      }
      #pragma unroll
      for (int i = 0; i < 4; ++i)
        #pragma unroll
        for (int j = 0; j < 4; ++j)
          acc[i][j] = __builtin_amdgcn_wmma_f32_16x16x32_bf16(
              /*neg_a=*/false, a[i].v, /*neg_b=*/false, b[j].v,
              /*c_mod=*/(short)0, acc[i][j], /*reuse_a=*/false, /*reuse_b=*/false);
    }
  }

  // C/D layout: VGPR r -> row (hi*8 + r), lane l15 -> column.
  #pragma unroll
  for (int i = 0; i < 4; ++i) {
    #pragma unroll
    for (int j = 0; j < 4; ++j) {
      const int col = colBase + j * 16 + l15;
      #pragma unroll
      for (int r = 0; r < 8; ++r) {
        const int row = rowBase + i * 16 + hi * 8 + r;
        C[(long long)row * ldC + col] = (__bf16)(acc[i][j][r] * scale);
      }
    }
  }
}

// ---------------------------------------------------------------------------
// Row softmax + argmax (first-max tie-break, like jnp.argmax), in place.
// One block per row; 256 threads x 8 contiguous bf16 each (M = 2048).
// hist[row] = seg_num - argmax (stored bf16 after the main output);
// browse = hist[0] < 4 stored right after the 1024 hist entries.
// ---------------------------------------------------------------------------
__global__ __launch_bounds__(256) void softmax_argmax_kernel(
    __bf16* __restrict__ scores, __bf16* __restrict__ hist,
    int Mcols, const int* __restrict__ seg_num_p)
{
  const int row = blockIdx.x;
  const int tid = threadIdx.x;
  __bf16* srow = scores + (long long)row * Mcols;

  bf16x8 raw = *reinterpret_cast<const bf16x8*>(srow + tid * 8);
  float v[8];
  float lmax = -3.0e38f; int lidx = 0;
  #pragma unroll
  for (int i = 0; i < 8; ++i) {
    v[i] = (float)raw[i];
    if (v[i] > lmax) { lmax = v[i]; lidx = tid * 8 + i; }  // strict > keeps first
  }

  __shared__ float smax[256];
  __shared__ int   sidx[256];
  __shared__ float ssum[256];
  smax[tid] = lmax; sidx[tid] = lidx;
  __syncthreads();
  for (int s = 128; s > 0; s >>= 1) {
    if (tid < s) {
      const float a = smax[tid], b = smax[tid + s];
      if (b > a || (b == a && sidx[tid + s] < sidx[tid])) {
        smax[tid] = b; sidx[tid] = sidx[tid + s];
      }
    }
    __syncthreads();
  }
  const float rmax = smax[0];
  const int   amax = sidx[0];

  float e[8];
  float lsum = 0.f;
  #pragma unroll
  for (int i = 0; i < 8; ++i) { e[i] = __expf(v[i] - rmax); lsum += e[i]; }
  ssum[tid] = lsum;
  __syncthreads();
  for (int s = 128; s > 0; s >>= 1) {
    if (tid < s) ssum[tid] += ssum[tid + s];
    __syncthreads();
  }
  const float inv = 1.0f / ssum[0];

  bf16x8 outv;
  #pragma unroll
  for (int i = 0; i < 8; ++i) outv[i] = (__bf16)(e[i] * inv);
  *reinterpret_cast<bf16x8*>(srow + tid * 8) = outv;

  if (tid == 0) {
    const int h = seg_num_p[0] - amax;
    hist[row] = (__bf16)(float)h;
    if (row == 0) hist[gridDim.x] = (__bf16)((h < 4) ? 1.0f : 0.0f);  // browse
  }
}

// ---------------------------------------------------------------------------
// bf16 transpose: in [bz][rows][cols] -> out [bz][cols][rows]. 32x32 LDS tile.
// ---------------------------------------------------------------------------
__global__ void transpose_bf16_kernel(const __bf16* __restrict__ in,
                                      __bf16* __restrict__ out,
                                      int rows, int cols)
{
  __shared__ __bf16 tile[32][33];
  const long long base = (long long)blockIdx.z * rows * cols;
  const int c0 = blockIdx.x * 32;
  const int r0 = blockIdx.y * 32;
  const int tx = threadIdx.x;   // 0..31
  const int ty = threadIdx.y;   // 0..7
  #pragma unroll
  for (int i = 0; i < 32; i += 8)
    tile[ty + i][tx] = in[base + (long long)(r0 + ty + i) * cols + (c0 + tx)];
  __syncthreads();
  #pragma unroll
  for (int i = 0; i < 32; i += 8)
    out[base + (long long)(c0 + ty + i) * rows + (r0 + tx)] = tile[tx][ty + i];
}

// ---------------------------------------------------------------------------
// Launch: DIM=4096, HIDDEN=4096, B=2, N=512, M=2048.
// ws layout: xq [1024,4096] bf16 (8MB) | mk [4096,4096] bf16 (32MB, reused as
// memT [B,4096,2048]) | scores [2,512,2048] bf16 (4MB, becomes probs in place).
// ---------------------------------------------------------------------------
extern "C" void kernel_launch(void* const* d_in, const int* in_sizes, int n_in,
                              void* d_out, int out_size, void* d_ws, size_t ws_size,
                              hipStream_t stream)
{
  (void)in_sizes; (void)n_in; (void)out_size; (void)ws_size;

  const __bf16* memory = (const __bf16*)d_in[0];   // [2, 2048, 4096]
  const __bf16* inputs = (const __bf16*)d_in[1];   // [2, 512, 4096]
  const __bf16* wq     = (const __bf16*)d_in[2];   // [4096, 4096]
  const __bf16* wk     = (const __bf16*)d_in[3];   // [4096, 4096]
  const int*    segp   = (const int*)d_in[4];      // scalar

  char* ws = (char*)d_ws;
  __bf16* xq   = (__bf16*)ws;                                            // 8 MB
  __bf16* mk   = (__bf16*)(ws + (size_t)1024 * 4096 * 2);                // 32 MB
  __bf16* sc   = (__bf16*)(ws + (size_t)1024 * 4096 * 2
                              + (size_t)4096 * 4096 * 2);                // 4 MB
  __bf16* memT = mk;  // mk's lifetime ends after the scores GEMM
  __bf16* out  = (__bf16*)d_out;
  __bf16* hist = out + (size_t)2 * 512 * 4096;     // 1024 entries + browse

  const dim3 blk(256);

  // 1) xq = inputs @ wq^T : [1024,4096] x [4096,4096]^T, K=4096
  gemm_bt_wmma_kernel<<<dim3(1024 / 256, 4096 / 128, 1), blk, 0, stream>>>(
      inputs, wq, xq, 4096, 4096, 4096, 4096, 0, 0, 0, 1.0f);

  // 2) mk = memory @ wk^T : [4096,4096] x [4096,4096]^T, K=4096
  gemm_bt_wmma_kernel<<<dim3(4096 / 256, 4096 / 128, 1), blk, 0, stream>>>(
      memory, wk, mk, 4096, 4096, 4096, 4096, 0, 0, 0, 1.0f);

  // 3) scores = (xq @ mk^T) / sqrt(4096), per batch: [512,4096] x [2048,4096]^T
  gemm_bt_wmma_kernel<<<dim3(512 / 256, 2048 / 128, 2), blk, 0, stream>>>(
      xq, mk, sc, 4096, 4096, 4096, 2048,
      (long long)512 * 4096, (long long)2048 * 4096, (long long)512 * 2048,
      0.015625f /* 1/64 */);

  // 4) softmax + argmax in place; hist/browse appended after main output
  softmax_argmax_kernel<<<dim3(1024), blk, 0, stream>>>(sc, hist, 2048, segp);

  // 5) memT[b][d][m] = memory[b][m][d]   (rows=2048, cols=4096 per batch)
  transpose_bf16_kernel<<<dim3(4096 / 32, 2048 / 32, 2), dim3(32, 8), 0, stream>>>(
      memory, memT, 2048, 4096);

  // 6) out = probs @ memT^T, per batch: [512,2048] x [4096,2048]^T, K=2048
  gemm_bt_wmma_kernel<<<dim3(512 / 256, 4096 / 128, 2), blk, 0, stream>>>(
      sc, memT, out, 2048, 2048, 2048, 4096,
      (long long)512 * 2048, (long long)4096 * 2048, (long long)512 * 4096,
      1.0f);
}
